// MHA_4818953306262
// MI455X (gfx1250) — compile-verified, matrix operand order checked by reading, not run on hardware
//
#include <hip/hip_runtime.h>
#include <hip/hip_bf16.h>

// ---------------------------------------------------------------------------
// B=8, S=2048, D_IN=1024, D_PROJ=256
//   x   = relu(src @ W^T + b)                    [B,S,256]  (fp16 in ws)
//   out = softmax(x x^T / 16) @ x                [B,S,256]  (fp32)
// Kernel 1: WMMA projection GEMM -> x (row major) and x^T, both f16 in ws.
// Kernel 2: fused flash attention; K/V tiles staged block-wide in LDS via
//           gfx1250 async global->LDS copies (ASYNCcnt), WMMA from LDS.
// ---------------------------------------------------------------------------

typedef _Float16 half_t;
typedef __attribute__((ext_vector_type(16))) _Float16 v16h;
typedef __attribute__((ext_vector_type(8)))  _Float16 v8h;
typedef __attribute__((ext_vector_type(8)))  float    v8f;
typedef __attribute__((ext_vector_type(4)))  float    v4f;

#define S_DIM 2048
#define K_DIM 1024
#define D_DIM 256
#define BATCH 8

__device__ __forceinline__ v8f wmma_f16(v16h a, v16h b, v8f c) {
    // (neg_a, A, neg_b, B, c_mod, C, reuse_a, reuse_b)
    return __builtin_amdgcn_wmma_f32_16x16x32_f16(false, a, false, b,
                                                  (short)0, c, false, false);
}

// Async 16B global->LDS copy (per-lane dest), tracked by ASYNCcnt.
__device__ __forceinline__ void async_copy_b128(unsigned lds_off,
                                                const void* gaddr) {
    asm volatile("global_load_async_to_lds_b128 %0, %1, off"
                 :: "v"(lds_off), "v"(gaddr) : "memory");
}

__device__ __forceinline__ void wait_asynccnt0() {
#if __has_builtin(__builtin_amdgcn_s_wait_asynccnt)
    __builtin_amdgcn_s_wait_asynccnt(0);
#else
    asm volatile("s_wait_asynccnt 0x0" ::: "memory");
#endif
}

// 16-bit A-fragment (16x32, M = lane%16) OR B-fragment (32x16, N = lane%16
// from column-major-equivalent rows): two contiguous 8-half chunks per lane
// at k0 + 8*(lane>=16) and +16.  `ld` in halves; keeps 16B alignment.
__device__ __forceinline__ v16h load_frag_f16(const half_t* __restrict__ p,
                                              int row0, int k0, int ld) {
    const int lane = threadIdx.x & 31;
    const half_t* q = p + (size_t)(row0 + (lane & 15)) * ld
                        + k0 + ((lane >> 4) << 3);
    union { v16h v; v8h h[2]; } u;
    u.h[0] = *(const v8h*)(q);
    u.h[1] = *(const v8h*)(q + 16);
    return u.v;
}

// Same fragment pattern from an f32 source, converting to f16 in registers.
__device__ __forceinline__ v16h load_frag_f32(const float* __restrict__ p,
                                              int row0, int k0, int ld) {
    const int lane = threadIdx.x & 31;
    const float* q = p + (size_t)(row0 + (lane & 15)) * ld
                       + k0 + ((lane >> 4) << 3);
    v4f a0 = *(const v4f*)(q);
    v4f a1 = *(const v4f*)(q + 4);
    v4f b0 = *(const v4f*)(q + 16);
    v4f b1 = *(const v4f*)(q + 20);
    v16h r;
#pragma unroll
    for (int i = 0; i < 4; ++i) {
        r[i]      = (_Float16)a0[i];
        r[4 + i]  = (_Float16)a1[i];
        r[8 + i]  = (_Float16)b0[i];
        r[12 + i] = (_Float16)b1[i];
    }
    return r;
}

// ---------------------------------------------------------------------------
// Kernel 1: x = relu(src @ W^T + b).  Grid 1024 x 128 threads (4 waves).
// Block handles a 16-row M tile; wave w handles N columns [64w, 64w+64).
// Writes x[b][s][d] (ld=256) and xt[b][d][s] (ld=2048), both f16.
// ---------------------------------------------------------------------------
__global__ void __launch_bounds__(128)
proj_relu_kernel(const float* __restrict__ src, const float* __restrict__ W,
                 const float* __restrict__ bias,
                 half_t* __restrict__ x, half_t* __restrict__ xt) {
    const int lane = threadIdx.x & 31;
    const int wave = threadIdx.x >> 5;
    const int m0   = blockIdx.x * 16;      // global row (b*S + s)
    const int n0   = wave * 64;

    v8f acc[4];
#pragma unroll
    for (int t = 0; t < 4; ++t)
#pragma unroll
        for (int r = 0; r < 8; ++r) acc[t][r] = 0.0f;

    for (int k0 = 0; k0 < K_DIM; k0 += 32) {
        v16h a = load_frag_f32(src, m0, k0, K_DIM);
#pragma unroll
        for (int t = 0; t < 4; ++t) {
            v16h bf = load_frag_f32(W, n0 + t * 16, k0, K_DIM);
            acc[t] = wmma_f16(a, bf, acc[t]);
        }
    }

    // Epilogue: bias + relu, store f16 x and x^T.
    const int col   = lane & 15;
    const int rbase = (lane >> 4) * 8;
    const int batch = m0 >> 11;            // S=2048 rows per batch
    const int sbase = (m0 & (S_DIM - 1)) + rbase;
    half_t* xtb = xt + (size_t)batch * D_DIM * S_DIM;
#pragma unroll
    for (int t = 0; t < 4; ++t) {
        const int n  = n0 + t * 16 + col;
        const float bv = bias[n];
#pragma unroll
        for (int r = 0; r < 8; ++r) {
            float v = acc[t][r] + bv;
            v = v > 0.0f ? v : 0.0f;
            half_t h = (half_t)v;
            x[(size_t)(m0 + rbase + r) * D_DIM + n] = h;
            xtb[(size_t)n * S_DIM + (sbase + r)]    = h;
        }
    }
}

// ---------------------------------------------------------------------------
// Kernel 2: fused attention.  Grid 256 x 128 threads (4 waves).
// Each wave owns a 16-query tile with full d=256.  All 4 waves share each
// 32-key chunk: the block async-copies K (32x256) and V^T (256x32) tiles
// into LDS once, then every wave builds its WMMA fragments with ds loads.
// ---------------------------------------------------------------------------
__global__ void __launch_bounds__(128)
attn_kernel(const half_t* __restrict__ x, const half_t* __restrict__ xt,
            float* __restrict__ out) {
    __shared__ half_t kbuf[32][256];       // K chunk, row-major [key][d]
    __shared__ half_t vbuf[256][32];       // V chunk, transposed [d][key]
    __shared__ half_t pbuf[4][16][40];     // per-wave P-transpose staging

    const int lane  = threadIdx.x & 31;
    const int wave  = threadIdx.x >> 5;
    const int tid   = threadIdx.x;
    const int bpb   = S_DIM / 64;          // 32 blocks per batch
    const int batch = blockIdx.x / bpb;
    const int q0    = (blockIdx.x % bpb) * 64 + wave * 16;

    const half_t* xb  = x  + (size_t)batch * S_DIM * D_DIM;
    const half_t* xtb = xt + (size_t)batch * D_DIM * S_DIM;

    const unsigned kbase = (unsigned)(uintptr_t)&kbuf[0][0];
    const unsigned vbase = (unsigned)(uintptr_t)&vbuf[0][0];

    // Q fragments: 16 rows x full d=256 (8 fragments of 16x32), kept live.
    v16h qf[8];
#pragma unroll
    for (int i = 0; i < 8; ++i) qf[i] = load_frag_f16(xb, q0, i * 32, D_DIM);

    v8f acc[16];
#pragma unroll
    for (int t = 0; t < 16; ++t)
#pragma unroll
        for (int r = 0; r < 8; ++r) acc[t][r] = 0.0f;

    float m_run[8], l_run[8];
#pragma unroll
    for (int r = 0; r < 8; ++r) { m_run[r] = -3.0e38f; l_run[r] = 0.0f; }

    const float scale = 0.0625f;           // 1/sqrt(256)
    const int col   = lane & 15;
    const int rbase = (lane >> 4) * 8;

    for (int kc = 0; kc < S_DIM; kc += 32) {
        // ---- stage K (32x256) and V^T (256x32) tiles into LDS ------------
        __syncthreads();                   // previous chunk fully consumed
#pragma unroll
        for (int i = 0; i < 8; ++i) {
            const int seg = i * 128 + tid;             // 0..1023, 16B each
            // K tile: 32 rows x 512B  (seg -> row = seg/32, 16B col = seg%32)
            const int kr = seg >> 5, ks = seg & 31;
            async_copy_b128(kbase + (unsigned)(kr * 256 + ks * 8) * 2,
                            xb + (size_t)(kc + kr) * D_DIM + ks * 8);
            // V tile: 256 rows x 64B  (seg -> row = seg/4, 16B col = seg%4)
            const int vr = seg >> 2, vs = seg & 3;
            async_copy_b128(vbase + (unsigned)(vr * 32 + vs * 8) * 2,
                            xtb + (size_t)vr * S_DIM + kc + vs * 8);
        }
        wait_asynccnt0();
        __syncthreads();                   // tiles visible to all waves

        // ---- scores: two 16x16 tiles, contraction over d=256 -------------
        v8f s0, s1;
#pragma unroll
        for (int r = 0; r < 8; ++r) { s0[r] = 0.0f; s1[r] = 0.0f; }
#pragma unroll
        for (int i = 0; i < 8; ++i) {
            v16h kf0 = load_frag_f16(&kbuf[0][0], 0,  i * 32, D_DIM);
            v16h kf1 = load_frag_f16(&kbuf[0][0], 16, i * 32, D_DIM);
            s0 = wmma_f16(qf[i], kf0, s0);
            s1 = wmma_f16(qf[i], kf1, s1);
        }

        // ---- online softmax (row reductions across 16 lanes) -------------
        float al[8];
#pragma unroll
        for (int r = 0; r < 8; ++r) {
            float a = s0[r] * scale;
            float b = s1[r] * scale;
            float cm = fmaxf(a, b);
            cm = fmaxf(cm, __shfl_xor(cm, 1, 32));
            cm = fmaxf(cm, __shfl_xor(cm, 2, 32));
            cm = fmaxf(cm, __shfl_xor(cm, 4, 32));
            cm = fmaxf(cm, __shfl_xor(cm, 8, 32));
            float nm = fmaxf(m_run[r], cm);
            al[r]    = __expf(m_run[r] - nm);
            m_run[r] = nm;
            float p0 = __expf(a - nm);
            float p1 = __expf(b - nm);
            s0[r] = p0; s1[r] = p1;
            float rs = p0 + p1;
            rs += __shfl_xor(rs, 1, 32);
            rs += __shfl_xor(rs, 2, 32);
            rs += __shfl_xor(rs, 4, 32);
            rs += __shfl_xor(rs, 8, 32);
            l_run[r] = l_run[r] * al[r] + rs;
        }

        // rescale running accumulators
#pragma unroll
        for (int t = 0; t < 16; ++t)
#pragma unroll
            for (int r = 0; r < 8; ++r) acc[t][r] *= al[r];

        // ---- transpose P (C-layout: lane=col) into A-layout via LDS ------
#pragma unroll
        for (int r = 0; r < 8; ++r) {
            pbuf[wave][rbase + r][col]      = (half_t)s0[r];
            pbuf[wave][rbase + r][16 + col] = (half_t)s1[r];
        }
        __syncthreads();
        v16h pf = load_frag_f16(&pbuf[wave][0][0], 0, 0, 40);

        // ---- out += P @ V  (V fragments from LDS V^T tile, ld=32) --------
#pragma unroll
        for (int t = 0; t < 16; ++t) {
            v16h vf = load_frag_f16(&vbuf[0][0], t * 16, 0, 32);
            acc[t] = wmma_f16(pf, vf, acc[t]);
        }
    }

    // ---- epilogue: divide by row sums, store fp32 ------------------------
    float inv[8];
#pragma unroll
    for (int r = 0; r < 8; ++r) inv[r] = 1.0f / l_run[r];
    float* ob = out + (size_t)batch * S_DIM * D_DIM;
#pragma unroll
    for (int t = 0; t < 16; ++t)
#pragma unroll
        for (int r = 0; r < 8; ++r)
            ob[(size_t)(q0 + rbase + r) * D_DIM + t * 16 + col] =
                acc[t][r] * inv[r];
}

// ---------------------------------------------------------------------------
extern "C" void kernel_launch(void* const* d_in, const int* in_sizes, int n_in,
                              void* d_out, int out_size, void* d_ws,
                              size_t ws_size, hipStream_t stream) {
    const float* src  = (const float*)d_in[0];   // [8,2048,1024]
    const float* W    = (const float*)d_in[1];   // [256,1024]
    const float* bias = (const float*)d_in[2];   // [256]
    float* out = (float*)d_out;                  // [8,2048,256]

    half_t* x  = (half_t*)d_ws;                          // 8 MB: [B*S][256]
    half_t* xt = x + (size_t)BATCH * S_DIM * D_DIM;      // 8 MB: [B][256][S]

    proj_relu_kernel<<<(BATCH * S_DIM) / 16, 128, 0, stream>>>(src, W, bias,
                                                               x, xt);
    attn_kernel<<<(BATCH * S_DIM) / 64, 128, 0, stream>>>(x, xt, out);
}
